// GraphEmbeddingLayer_69518340653483
// MI455X (gfx1250) — matrix operand order repaired and checked by reading, not compiled, not run on hardware
//
#include <hip/hip_runtime.h>
#include <hip/hip_bf16.h>

// ---------------------------------------------------------------------------
// GAT layer for MI455X (gfx1250, wave32, WMMA).
//   x   = emb @ W                      (bf16 WMMA, f32 accum)
//   edge softmax + scatter             (atomic f32, edge-parallel waves)
//   out = (agg + bias) @ W_out + b_out (bf16 WMMA, f32 accum)
// ---------------------------------------------------------------------------

typedef __attribute__((ext_vector_type(16))) __bf16          v16bf;
typedef __attribute__((ext_vector_type(16))) unsigned short  v16u;
typedef __attribute__((ext_vector_type(8)))  float           v8f;

#define HDIM 4
#define CDIM 128
#define HC   512

__device__ __forceinline__ unsigned short f2bf(float f) {
    unsigned u = __float_as_uint(f);
    u += 0x7FFFu + ((u >> 16) & 1u);     // round-to-nearest-even
    return (unsigned short)(u >> 16);
}

// monotonic float <-> uint encoding so atomicMax(uint) == float max
__device__ __forceinline__ unsigned enc_f(float f) {
    unsigned u = __float_as_uint(f);
    return (u & 0x80000000u) ? ~u : (u | 0x80000000u);
}
__device__ __forceinline__ float dec_f(unsigned u) {
    u = (u & 0x80000000u) ? (u & 0x7FFFFFFFu) : ~u;
    return __uint_as_float(u);
}

__device__ __forceinline__ float lrelu(float a) { return a > 0.f ? a : 0.2f * a; }

// ---------------------------------------------------------------------------
// Tiled bf16-WMMA GEMM: Cout[M,Ncol] = (A[M,K] + a_bias) @ B[K,Ncol] + c_bias
// Block: 256 threads = 8 waves. Block tile 64(M) x 128(N), K-chunk 32.
// Each wave owns a 16-col stripe with 4 m-subtile accumulators; one B fragment
// feeds 4 v_wmma per K-chunk.
// LDS tiles live in per-lane FRAGMENT ORDER:
//   A elem (r,c): lane=(r&15)+16*((c>>3)&1), e=((c>>4)&1)<<3 | ((c>>1)&3)<<1 | (c&1)
//     -> 4 consecutive c in an aligned 8-block = 4 consecutive e  => ds_store_b64
//   B elem (k,c): grp=c>>4, lane=(c&15)+16*((k>>3)&1), e=((k>>4)&1)<<3 | (k&7)
//     -> (k,k+1) pair for a col packs into one dword              => ds_store_b32
// Fragment read = one contiguous 32B per lane (2x ds_load_b128).
// ---------------------------------------------------------------------------
template <bool ABIAS, bool CBIAS>
__global__ __launch_bounds__(256)
void gemm_bf16_wmma(const float* __restrict__ A, const float* __restrict__ B,
                    float* __restrict__ Cout, int M, int K, int Ncol,
                    const float* __restrict__ a_bias, const float* __restrict__ c_bias)
{
    __shared__ __align__(32) unsigned short sAf[4 * 512];   // 4 m-subtiles
    __shared__ __align__(32) unsigned short sBf[8 * 512];   // 8 col-groups

    const int t    = threadIdx.x;
    const int wave = t >> 5;
    const int lane = t & 31;
    const int m0   = blockIdx.x * 64;
    const int n0   = blockIdx.y * 128;
    const int half = lane >> 4;
    const int rr   = lane & 15;

    v8f acc[4] = {};

    for (int k0 = 0; k0 < K; k0 += 32) {
        // ---- stage A tile (64 rows x 32 K): float4 load -> ds_store_b64 ----
#pragma unroll
        for (int i = 0; i < 2; ++i) {
            int q = t + i * 256;                 // quad id 0..511
            int r = q >> 3, c = (q & 7) << 2;    // r 0..63, c in {0,4,...,28}
            int mr = m0 + r; if (mr > M - 1) mr = M - 1;
            float4 v = *(const float4*)&A[(size_t)mr * K + (k0 + c)];
            if (ABIAS) {
                const float4 bv = *(const float4*)&a_bias[k0 + c];
                v.x += bv.x; v.y += bv.y; v.z += bv.z; v.w += bv.w;
            }
            int hf = (c >> 3) & 1;
            int e0 = (((c >> 4) & 1) << 3) | (((c >> 1) & 3) << 1);
            int ofs = (r >> 4) * 512 + (((r & 15) + 16 * hf) << 4) + e0;
            uint2 pk;
            pk.x = (unsigned)f2bf(v.x) | ((unsigned)f2bf(v.y) << 16);
            pk.y = (unsigned)f2bf(v.z) | ((unsigned)f2bf(v.w) << 16);
            *(uint2*)&sAf[ofs] = pk;
        }
        // ---- stage B tile (32 K x 128 cols): 2 rows x 4 cols -> 4x ds_store_b32 ----
#pragma unroll
        for (int i = 0; i < 2; ++i) {
            int item = t + i * 256;              // 0..511: kp = item>>5, cg = item&31
            int k = (item >> 5) << 1, c = (item & 31) << 2;
            const float* b0 = &B[(size_t)(k0 + k) * Ncol + (n0 + c)];
            float4 ra = *(const float4*)b0;
            float4 rb = *(const float4*)(b0 + Ncol);
            float va[4] = {ra.x, ra.y, ra.z, ra.w};
            float vb[4] = {rb.x, rb.y, rb.z, rb.w};
#pragma unroll
            for (int j = 0; j < 4; ++j) {
                int cj = c + j;
                unsigned pk = (unsigned)f2bf(va[j]) | ((unsigned)f2bf(vb[j]) << 16);
                int ofs = (cj >> 4) * 512 + (((cj & 15) + 16 * ((k >> 3) & 1)) << 4) + ((((k >> 4) & 1) << 3) | (k & 7));
                *(unsigned*)&sBf[ofs] = pk;
            }
        }
        if (k0 + 32 < K)   // pull next B tile toward L2 (global_prefetch_b8)
            __builtin_prefetch(&B[(size_t)(k0 + 32) * Ncol + n0 + (t & 127)], 0, 0);
        __syncthreads();

        // ---- fragments: contiguous 32B per lane ----
        v16u bu = *(const v16u*)&sBf[(wave << 9) + (lane << 4)];
        v16bf bfr = __builtin_bit_cast(v16bf, bu);
#pragma unroll
        for (int s = 0; s < 4; ++s) {
            v16u au = *(const v16u*)&sAf[(s << 9) + (lane << 4)];
            acc[s] = __builtin_amdgcn_wmma_f32_16x16x32_bf16(
                         false, __builtin_bit_cast(v16bf, au), false, bfr,
                         (short)0, acc[s], false, false);
        }
        __syncthreads();
    }

    // D layout: lanes 0-15 -> rows 0..7 (VGPR r), lanes 16-31 -> rows 8..15; col = rr
    const int col = n0 + wave * 16 + rr;
    float cb = CBIAS ? c_bias[col] : 0.f;
#pragma unroll
    for (int s = 0; s < 4; ++s) {
#pragma unroll
        for (int r = 0; r < 8; ++r) {
            int m = m0 + s * 16 + r + half * 8;
            if (m < M) Cout[(size_t)m * Ncol + col] = acc[s][r] + cb;
        }
    }
}

// ---------------------------------------------------------------------------
__global__ void fill_u32(unsigned* __restrict__ p, unsigned v, size_t n) {
    size_t i = (size_t)blockIdx.x * blockDim.x + threadIdx.x;
    if (i < n) p[i] = v;
}

// a_src[n,h] = sum_c x[n,h,c]*att_src[h,c]; same for a_dst. One wave per (n,h).
__global__ __launch_bounds__(256)
void node_attn(const float* __restrict__ x, const float* __restrict__ att_src,
               const float* __restrict__ att_dst, float* __restrict__ a_src,
               float* __restrict__ a_dst, int Nn)
{
    int gw   = (int)((blockIdx.x * blockDim.x + threadIdx.x) >> 5);
    int lane = threadIdx.x & 31;
    if (gw >= Nn * HDIM) return;
    int n = gw >> 2, h = gw & 3;
    const float* xr = x + (size_t)n * HC + h * CDIM;
    float s = 0.f, d = 0.f;
    for (int c = lane; c < CDIM; c += 32) {
        float xv = xr[c];
        s += xv * att_src[h * CDIM + c];
        d += xv * att_dst[h * CDIM + c];
    }
    for (int off = 16; off; off >>= 1) {
        s += __shfl_down(s, off, 32);
        d += __shfl_down(d, off, 32);
    }
    if (lane == 0) { a_src[gw] = s; a_dst[gw] = d; }
}

__global__ void degree_k(const int* __restrict__ dst, const float* __restrict__ ew,
                         float* __restrict__ cnt, float* __restrict__ wsum, int E)
{
    int i = blockIdx.x * blockDim.x + threadIdx.x;
    if (i >= E) return;
    int d = dst[i];
    atomicAdd(&cnt[d], 1.0f);
    atomicAdd(&wsum[d], ew[i]);
}

__global__ void mean_k(const float* __restrict__ cnt, float* __restrict__ wsum, int Nn) {
    int i = blockIdx.x * blockDim.x + threadIdx.x;
    if (i < Nn) wsum[i] = wsum[i] / fmaxf(cnt[i], 1.0f);   // in-place -> ea_mean
}

// k[h] = sum_c att_edge[h,c] * W_edge[h,c]   (e_feat = ea outer W_edge collapses)
__global__ void ke_k(const float* __restrict__ att_edge, const float* __restrict__ W_edge,
                     float* __restrict__ ke)
{
    int h = threadIdx.x >> 5, lane = threadIdx.x & 31;
    float s = 0.f;
    for (int c = lane; c < CDIM; c += 32)
        s += att_edge[h * CDIM + c] * W_edge[h * CDIM + c];
    for (int off = 16; off; off >>= 1) s += __shfl_down(s, off, 32);
    if (lane == 0) ke[h] = s;
}

__device__ __forceinline__ void item_edge(int i, int E, const int* src, const int* dst,
                                          const float* ew, const float* ea_mean,
                                          int& s, int& d, float& w)
{
    if (i < E) { s = src[i]; d = dst[i]; w = ew[i]; }
    else       { s = d = i - E; w = ea_mean[i - E]; }
}

__global__ void max_pass(const int* __restrict__ src, const int* __restrict__ dst,
                         const float* __restrict__ ew, const float* __restrict__ ea_mean,
                         const float* __restrict__ a_src, const float* __restrict__ a_dst,
                         const float* __restrict__ ke, unsigned* __restrict__ m_enc,
                         int E, int Nn)
{
    int i = blockIdx.x * blockDim.x + threadIdx.x;
    if (i >= E + Nn) return;
    int s, d; float w;
    item_edge(i, E, src, dst, ew, ea_mean, s, d, w);
#pragma unroll
    for (int h = 0; h < HDIM; ++h) {
        float a = lrelu(a_src[s * HDIM + h] + a_dst[d * HDIM + h] + w * ke[h]);
        atomicMax(&m_enc[d * HDIM + h], enc_f(a));
    }
}

__global__ void denom_pass(const int* __restrict__ src, const int* __restrict__ dst,
                           const float* __restrict__ ew, const float* __restrict__ ea_mean,
                           const float* __restrict__ a_src, const float* __restrict__ a_dst,
                           const float* __restrict__ ke, const unsigned* __restrict__ m_enc,
                           float* __restrict__ denom, int E, int Nn)
{
    int i = blockIdx.x * blockDim.x + threadIdx.x;
    if (i >= E + Nn) return;
    int s, d; float w;
    item_edge(i, E, src, dst, ew, ea_mean, s, d, w);
#pragma unroll
    for (int h = 0; h < HDIM; ++h) {
        float a = lrelu(a_src[s * HDIM + h] + a_dst[d * HDIM + h] + w * ke[h]);
        atomicAdd(&denom[d * HDIM + h], __expf(a - dec_f(m_enc[d * HDIM + h])));
    }
}

// One block (128 thr = 4 waves) per edge item; wave h scatters head h (128 ch, float4/lane).
__global__ __launch_bounds__(128)
void scatter_pass(const int* __restrict__ src, const int* __restrict__ dst,
                  const float* __restrict__ ew, const float* __restrict__ ea_mean,
                  const float* __restrict__ a_src, const float* __restrict__ a_dst,
                  const float* __restrict__ ke, const unsigned* __restrict__ m_enc,
                  const float* __restrict__ denom, const float* __restrict__ x,
                  float* __restrict__ agg, int E, int Nn)
{
    int i = blockIdx.x;
    int s, d; float w;
    item_edge(i, E, src, dst, ew, ea_mean, s, d, w);
    int h = threadIdx.x >> 5, lane = threadIdx.x & 31;
    float a  = lrelu(a_src[s * HDIM + h] + a_dst[d * HDIM + h] + w * ke[h]);
    float wt = __expf(a - dec_f(m_enc[d * HDIM + h])) / (denom[d * HDIM + h] + 1e-16f);
    const float4* xs = (const float4*)(x + (size_t)s * HC + h * CDIM);
    float* ag = agg + (size_t)d * HC + h * CDIM + lane * 4;
    float4 v = xs[lane];
    atomicAdd(&ag[0], v.x * wt);
    atomicAdd(&ag[1], v.y * wt);
    atomicAdd(&ag[2], v.z * wt);
    atomicAdd(&ag[3], v.w * wt);
}

// ---------------------------------------------------------------------------
extern "C" void kernel_launch(void* const* d_in, const int* in_sizes, int n_in,
                              void* d_out, int out_size, void* d_ws, size_t ws_size,
                              hipStream_t stream)
{
    const int*   ei       = (const int*)d_in[0];
    const float* ew       = (const float*)d_in[1];
    const float* emb      = (const float*)d_in[2];
    const float* W        = (const float*)d_in[3];
    const float* att_src  = (const float*)d_in[4];
    const float* att_dst  = (const float*)d_in[5];
    const float* att_edge = (const float*)d_in[6];
    const float* W_edge   = (const float*)d_in[7];
    const float* bias     = (const float*)d_in[8];
    const float* W_out    = (const float*)d_in[9];
    const float* b_out    = (const float*)d_in[10];
    float*       out      = (float*)d_out;

    const int Dm = 256;
    const int E  = in_sizes[1];
    const int Nn = in_sizes[2] / Dm;
    const int EN = E + Nn;
    const int* src = ei;
    const int* dst = ei + E;

    // workspace carve-out
    float* ws = (float*)d_ws;
    size_t off = 0;
    float* x      = ws + off; off += (size_t)Nn * HC;
    float* agg    = ws + off; off += (size_t)Nn * HC;
    float* a_src  = ws + off; off += (size_t)Nn * HDIM;
    float* a_dst  = ws + off; off += (size_t)Nn * HDIM;
    float* zbase  = ws + off;                      // zero-init region starts here
    float* cnt    = ws + off; off += Nn;
    float* wsum   = ws + off; off += Nn;           // becomes ea_mean in place
    unsigned* m_enc = (unsigned*)(ws + off); off += (size_t)Nn * HDIM;
    float* denom  = ws + off; off += (size_t)Nn * HDIM;
    float* ke     = ws + off; off += HDIM;

    // init: zero agg and [cnt..denom] (enc_f minimum == 0)
    {
        size_t n1 = (size_t)Nn * HC;
        fill_u32<<<dim3((unsigned)((n1 + 255) / 256)), 256, 0, stream>>>((unsigned*)agg, 0u, n1);
        size_t n2 = (size_t)Nn * (1 + 1 + HDIM + HDIM);
        fill_u32<<<dim3((unsigned)((n2 + 255) / 256)), 256, 0, stream>>>((unsigned*)zbase, 0u, n2);
    }

    // 1) x = emb @ W   [Nn,256] x [256,512]
    gemm_bf16_wmma<false, false><<<dim3((Nn + 63) / 64, HC / 128), 256, 0, stream>>>(
        emb, W, x, Nn, Dm, HC, nullptr, nullptr);

    // 2) per-node attention dots
    node_attn<<<(Nn * HDIM + 7) / 8, 256, 0, stream>>>(x, att_src, att_dst, a_src, a_dst, Nn);

    // 3) degrees + per-dst mean edge weight (self-loop attr)
    degree_k<<<(E + 255) / 256, 256, 0, stream>>>(dst, ew, cnt, wsum, E);
    mean_k<<<(Nn + 255) / 256, 256, 0, stream>>>(cnt, wsum, Nn);

    // 4) k[h] = <att_edge[h], W_edge[h]>
    ke_k<<<1, HDIM * 32, 0, stream>>>(att_edge, W_edge, ke);

    // 5-7) segment softmax + weighted scatter
    max_pass<<<(EN + 255) / 256, 256, 0, stream>>>(src, dst, ew, wsum, a_src, a_dst, ke,
                                                   m_enc, E, Nn);
    denom_pass<<<(EN + 255) / 256, 256, 0, stream>>>(src, dst, ew, wsum, a_src, a_dst, ke,
                                                     m_enc, denom, E, Nn);
    scatter_pass<<<EN, 128, 0, stream>>>(src, dst, ew, wsum, a_src, a_dst, ke,
                                         m_enc, denom, x, agg, E, Nn);

    // 8) out = (agg + bias) @ W_out + b_out   [Nn,512] x [512,256]
    gemm_bf16_wmma<true, true><<<dim3((Nn + 63) / 64, Dm / 128), 256, 0, stream>>>(
        agg, W_out, out, Nn, HC, Dm, bias, b_out);
}
